// CrossAttention_4320737099861
// MI455X (gfx1250) — matrix operand order repaired and checked by reading, not compiled, not run on hardware
//
#include <hip/hip_runtime.h>
#include <cmath>

typedef _Float16 half_t;
typedef __attribute__((ext_vector_type(4)))  _Float16 v4h;
typedef __attribute__((ext_vector_type(8)))  _Float16 v8h;
typedef __attribute__((ext_vector_type(16))) _Float16 v16h;
typedef __attribute__((ext_vector_type(4)))  float    v4f;
typedef __attribute__((ext_vector_type(8)))  float    v8f;

constexpr int B_ = 8;
constexpr int L_ = 2048;
constexpr int H_ = 1024;
constexpr float INV_SCALE = 1.0f / 32.0f;   // 1/sqrt(1024)
constexpr float EPS_ = 1e-5f;

// ---------------------------------------------------------------------------
// WMMA helpers (CDNA5 16x16x32 f16 -> f32)
// ---------------------------------------------------------------------------
__device__ __forceinline__ v8f wmma16x16x32(v16h a, v16h b, v8f c) {
  return __builtin_amdgcn_wmma_f32_16x16x32_f16(false, a, false, b, (short)0, c,
                                                false, false);
}

// A-matrix 16x32 f16 fragment from a row-major buffer (ld in halves).
__device__ __forceinline__ v16h frag_a(const half_t* base, int ld, int lane) {
  const int row = lane & 15, hs = lane >> 4;
  const half_t* p = base + row * ld + hs * 16;
  v8h lo = *(const v8h*)(p);
  v8h hi = *(const v8h*)(p + 8);
  v16h r;
#pragma unroll
  for (int i = 0; i < 8; ++i) { r[i] = lo[i]; r[i + 8] = hi[i]; }
  return r;
}

// B-matrix 32x16 f16 fragment; each column's K values contiguous (ldc in halves).
__device__ __forceinline__ v16h frag_b(const half_t* base, int ldc, int lane) {
  const int col = lane & 15, koff = (lane >> 4) * 16;
  return *(const v16h*)(base + (size_t)col * ldc + koff);
}

// ---------------------------------------------------------------------------
// Kernel 1: C = X @ W + bias, f16 output. STORE_VT=1 stores transposed [B][H][L].
// Block tile 128x128, 8 waves (each 32x64), BK=32.
// ---------------------------------------------------------------------------
template <int STORE_VT>
__global__ __launch_bounds__(256)
void proj_gemm_f16(const float* __restrict__ X, const float* __restrict__ W,
                   const float* __restrict__ bias, half_t* __restrict__ out) {
  __shared__ __align__(32) half_t lx[128 * 32];   // [m][k]
  __shared__ __align__(32) half_t lw[128 * 32];   // [n][k] (transposed W tile)

  const int tid  = threadIdx.x;
  const int lane = tid & 31;
  const int w    = tid >> 5;
  const int wm   = w >> 1;   // 0..3
  const int wn   = w & 1;    // 0..1
  const int n0   = blockIdx.x * 128;
  const int m0   = blockIdx.y * 128;

  v8f acc[2][4] = {};

  for (int k0 = 0; k0 < H_; k0 += 32) {
    // X tile: 1024 float4 loads -> packed f16 b64 LDS stores
#pragma unroll
    for (int t = 0; t < 4; ++t) {
      int q = tid + t * 256;                 // quad index
      int m = q >> 3, k4 = (q & 7) * 4;
      v4f xv = *(const v4f*)&X[(size_t)(m0 + m) * H_ + (k0 + k4)];
      v4h hv;
#pragma unroll
      for (int i = 0; i < 4; ++i) hv[i] = (half_t)xv[i];
      *(v4h*)&lx[m * 32 + k4] = hv;
    }
    // W tile: coalesced float4 along n, scattered transposed b16 LDS stores
#pragma unroll
    for (int t = 0; t < 4; ++t) {
      int q = tid + t * 256;
      int k = q >> 5, n4 = (q & 31) * 4;
      v4f wv = *(const v4f*)&W[(size_t)(k0 + k) * H_ + (n0 + n4)];
#pragma unroll
      for (int i = 0; i < 4; ++i) lw[(n4 + i) * 32 + k] = (half_t)wv[i];
    }
    __syncthreads();

    v16h bf[4];
#pragma unroll
    for (int nt = 0; nt < 4; ++nt)
      bf[nt] = frag_b(&lw[(wn * 64 + nt * 16) * 32], 32, lane);
#pragma unroll
    for (int mt = 0; mt < 2; ++mt) {
      v16h a = frag_a(&lx[(wm * 32 + mt * 16) * 32], 32, lane);
#pragma unroll
      for (int nt = 0; nt < 4; ++nt)
        acc[mt][nt] = wmma16x16x32(a, bf[nt], acc[mt][nt]);
    }
    __syncthreads();
  }

  const int hs = lane >> 4, cn = lane & 15;
#pragma unroll
  for (int mt = 0; mt < 2; ++mt)
#pragma unroll
    for (int nt = 0; nt < 4; ++nt)
#pragma unroll
      for (int vv = 0; vv < 8; ++vv) {
        int m = m0 + wm * 32 + mt * 16 + vv + 8 * hs;
        int n = n0 + wn * 64 + nt * 16 + cn;
        half_t hv = (half_t)(acc[mt][nt][vv] + bias[n]);
        if (STORE_VT) {
          int bb = m >> 11, ml = m & (L_ - 1);           // L_ = 2048
          out[((size_t)bb * H_ + n) * L_ + ml] = hv;     // [B][H][L]
        } else {
          out[(size_t)m * H_ + n] = hv;                  // [B*L][H]
        }
      }
}

// ---------------------------------------------------------------------------
// Kernel 2: flash attention (masked, online softmax) + LayerNorm, fused.
// One block per (batch, 32-query tile); 8 waves; wave w owns H cols [w*128,+128).
// B-fragment loads are software-pipelined one step ahead of the WMMAs so the
// L2 load latency overlaps matrix math instead of stalling at s_wait_loadcnt.
// ---------------------------------------------------------------------------
__global__ __launch_bounds__(256)
void attn_ln_kernel(const half_t* __restrict__ Qh, const half_t* __restrict__ Kh,
                    const half_t* __restrict__ Vt, const int* __restrict__ mask,
                    const float* __restrict__ gamma, const float* __restrict__ beta,
                    float* __restrict__ out) {
  __shared__ __align__(32) half_t sQ[32 * 1024];   // 64 KB  [r][h]
  __shared__ __align__(32) float  sS[32 * 256];    // 32 KB  raw/masked scores
  __shared__ __align__(32) half_t sP[32 * 256];    // 16 KB  probabilities f16
  __shared__ float sM[32], sL[32], sAl[32], sSum[32], sSq[32];

  const int tid  = threadIdx.x;
  const int lane = tid & 31;
  const int w    = tid >> 5;
  const int q0   = blockIdx.x * 32;
  const int b    = blockIdx.y;
  const int hs   = lane >> 4, cn = lane & 15;

  // stage Q tile (32 x H) into LDS
#pragma unroll
  for (int t = 0; t < 16; ++t) {
    int chunk = tid + t * 256;                 // 4096 chunks of 8 halves
    int r = chunk >> 7, c8 = (chunk & 127) * 8;
    *(v8h*)&sQ[r * 1024 + c8] =
        *(const v8h*)&Qh[((size_t)(b * L_) + q0 + r) * H_ + c8];
  }
  if (tid < 32) { sM[tid] = -INFINITY; sL[tid] = 0.0f; sSum[tid] = 0.0f; sSq[tid] = 0.0f; }

  v8f acc[2][8] = {};
  __syncthreads();

  const half_t* kbase0 = Kh + ((size_t)(b * L_) + w * 32) * H_;
  const half_t* vbase0 = Vt + ((size_t)b * H_ + w * 128) * L_;

  for (int kb = 0; kb < L_; kb += 256) {
    const half_t* kbase = kbase0 + (size_t)kb * H_;
    const half_t* vbase = vbase0 + kb;

    // warm L2/WGP$ for the next key block (global_prefetch_b8)
    if (kb + 256 < L_) {
      __builtin_prefetch(kbase + (size_t)256 * H_ + (size_t)lane * H_, 0, 1);
      __builtin_prefetch(vbase + 256 + (size_t)(lane * 4) * L_, 0, 1);
    }

    // ---- S = Q @ K^T : wave w computes 32 rows x keys [kb+w*32, +32) ----
    // Pipeline the two K B-fragments one kc-step ahead of the WMMAs.
    v8f sc[2][2] = {};
    {
      v16h b0 = frag_b(kbase, H_, lane);
      v16h b1 = frag_b(kbase + (size_t)16 * H_, H_, lane);
#pragma unroll
      for (int kc = 0; kc < H_ / 32; ++kc) {
        v16h b0n = b0, b1n = b1;
        if (kc + 1 < H_ / 32) {
          b0n = frag_b(kbase + (kc + 1) * 32, H_, lane);
          b1n = frag_b(kbase + (size_t)16 * H_ + (kc + 1) * 32, H_, lane);
        }
        v16h a0 = frag_a(&sQ[kc * 32], 1024, lane);
        v16h a1 = frag_a(&sQ[16 * 1024 + kc * 32], 1024, lane);
        sc[0][0] = wmma16x16x32(a0, b0, sc[0][0]);
        sc[1][0] = wmma16x16x32(a1, b0, sc[1][0]);
        sc[0][1] = wmma16x16x32(a0, b1, sc[0][1]);
        sc[1][1] = wmma16x16x32(a1, b1, sc[1][1]);
        b0 = b0n; b1 = b1n;
      }
    }
#pragma unroll
    for (int mt = 0; mt < 2; ++mt)
#pragma unroll
      for (int kt = 0; kt < 2; ++kt)
#pragma unroll
        for (int vv = 0; vv < 8; ++vv)
          sS[(mt * 16 + vv + 8 * hs) * 256 + w * 32 + kt * 16 + cn] = sc[mt][kt][vv];
    __syncthreads();

    // ---- masked online softmax: 8 threads per query row, 2 passes over LDS ----
    {
      const int r = tid >> 3, j0 = tid & 7;
      const int* mrow = mask + ((size_t)b * L_ + q0 + r) * L_ + kb;
      float mloc = -INFINITY;
#pragma unroll
      for (int jj = 0; jj < 32; ++jj) {          // pass 1: mask+scale in place
        int j = j0 * 32 + jj;
        float s = sS[r * 256 + j] * INV_SCALE;
        if (mrow[j] == 0) s = -INFINITY;
        sS[r * 256 + j] = s;
        mloc = fmaxf(mloc, s);
      }
#pragma unroll
      for (int off = 4; off >= 1; off >>= 1)
        mloc = fmaxf(mloc, __shfl_xor(mloc, off, 32));
      float mold = sM[r];
      float mnew = fmaxf(mold, mloc);
      float alpha = (mold > -INFINITY) ? __expf(mold - mnew) : 0.0f;
      float lsum = 0.0f;
#pragma unroll
      for (int jj = 0; jj < 32; ++jj) {          // pass 2: exponentiate
        int j = j0 * 32 + jj;
        float s = sS[r * 256 + j];
        float p = (s > -INFINITY) ? __expf(s - mnew) : 0.0f;
        lsum += p;
        sP[r * 256 + j] = (half_t)p;
      }
#pragma unroll
      for (int off = 4; off >= 1; off >>= 1)
        lsum += __shfl_xor(lsum, off, 32);
      if (j0 == 0) { sL[r] = sL[r] * alpha + lsum; sM[r] = mnew; sAl[r] = alpha; }
    }
    __syncthreads();

    // ---- rescale accumulators, acc += P @ V (Vt layout [B][H][L]) ----
    float al[2][8];
#pragma unroll
    for (int mt = 0; mt < 2; ++mt)
#pragma unroll
      for (int vv = 0; vv < 8; ++vv) al[mt][vv] = sAl[mt * 16 + vv + 8 * hs];
#pragma unroll
    for (int mt = 0; mt < 2; ++mt)
#pragma unroll
      for (int nt = 0; nt < 8; ++nt)
#pragma unroll
        for (int vv = 0; vv < 8; ++vv) acc[mt][nt][vv] *= al[mt][vv];

#pragma unroll
    for (int kc = 0; kc < 8; ++kc) {
      v16h a0 = frag_a(&sP[kc * 32], 256, lane);
      v16h a1 = frag_a(&sP[16 * 256 + kc * 32], 256, lane);
      v16h bb = frag_b(vbase + kc * 32, L_, lane);     // nt = 0
#pragma unroll
      for (int nt = 0; nt < 8; ++nt) {
        v16h bbn = bb;
        if (nt + 1 < 8)
          bbn = frag_b(vbase + (size_t)(nt + 1) * 16 * L_ + kc * 32, L_, lane);
        acc[0][nt] = wmma16x16x32(a0, bb, acc[0][nt]);
        acc[1][nt] = wmma16x16x32(a1, bb, acc[1][nt]);
        bb = bbn;
      }
    }
    __syncthreads();
  }

  // ---- epilogue: divide by softmax denom, LayerNorm across waves ----
#pragma unroll
  for (int mt = 0; mt < 2; ++mt)
#pragma unroll
    for (int vv = 0; vv < 8; ++vv) {
      float linv = 1.0f / sL[mt * 16 + vv + 8 * hs];
      float s = 0.0f, q = 0.0f;
#pragma unroll
      for (int nt = 0; nt < 8; ++nt) {
        float v = acc[mt][nt][vv] * linv;
        acc[mt][nt][vv] = v;
        s += v; q += v * v;
      }
#pragma unroll
      for (int off = 8; off >= 1; off >>= 1) {
        s += __shfl_xor(s, off, 32);
        q += __shfl_xor(q, off, 32);
      }
      if (cn == 0) {
        atomicAdd(&sSum[mt * 16 + vv + 8 * hs], s);
        atomicAdd(&sSq[mt * 16 + vv + 8 * hs], q);
      }
    }
  __syncthreads();
  if (tid < 32) {
    float mu  = sSum[tid] * (1.0f / H_);
    float var = sSq[tid] * (1.0f / H_) - mu * mu;
    sSum[tid] = mu;
    sSq[tid]  = rsqrtf(var + EPS_);
  }
  __syncthreads();
#pragma unroll
  for (int mt = 0; mt < 2; ++mt)
#pragma unroll
    for (int nt = 0; nt < 8; ++nt)
#pragma unroll
      for (int vv = 0; vv < 8; ++vv) {
        int row = mt * 16 + vv + 8 * hs;
        int col = w * 128 + nt * 16 + cn;
        float y = (acc[mt][nt][vv] - sSum[row]) * sSq[row] * gamma[col] + beta[col];
        out[((size_t)(b * L_) + q0 + row) * H_ + col] = y;
      }
}

// ---------------------------------------------------------------------------
extern "C" void kernel_launch(void* const* d_in, const int* in_sizes, int n_in,
                              void* d_out, int out_size, void* d_ws, size_t ws_size,
                              hipStream_t stream) {
  (void)in_sizes; (void)n_in; (void)out_size; (void)ws_size;
  const float* queries = (const float*)d_in[0];
  const float* keys    = (const float*)d_in[1];
  const float* values  = (const float*)d_in[2];
  const int*   mask    = (const int*)d_in[3];
  const float* Wq = (const float*)d_in[4];
  const float* bq = (const float*)d_in[5];
  const float* Wk = (const float*)d_in[6];
  const float* bk = (const float*)d_in[7];
  const float* Wv = (const float*)d_in[8];
  const float* bv = (const float*)d_in[9];
  const float* g  = (const float*)d_in[10];
  const float* be = (const float*)d_in[11];
  float* out = (float*)d_out;

  const size_t elems = (size_t)B_ * L_ * H_;   // 16.7M -> 32 MB f16 each
  half_t* Qh = (half_t*)d_ws;
  half_t* Kh = Qh + elems;
  half_t* Vt = Kh + elems;                     // stored [B][H][L]

  dim3 blk(256);
  dim3 ggrid(H_ / 128, (B_ * L_) / 128);
  proj_gemm_f16<0><<<ggrid, blk, 0, stream>>>(queries, Wq, bq, Qh);
  proj_gemm_f16<0><<<ggrid, blk, 0, stream>>>(keys,    Wk, bk, Kh);
  proj_gemm_f16<1><<<ggrid, blk, 0, stream>>>(values,  Wv, bv, Vt);

  dim3 agrid(L_ / 32, B_);
  attn_ln_kernel<<<agrid, blk, 0, stream>>>(Qh, Kh, Vt, mask, g, be, out);
}
